// Attention_58652073394851
// MI455X (gfx1250) — matrix operand order repaired and checked by reading, not compiled, not run on hardware
//
#include <hip/hip_runtime.h>
#include <hip/hip_bf16.h>
#include <math.h>

// Sizes from the reference: D=256 (input), H=64 (hidden). N inferred at launch.
#define DD 256
#define HH 64
#define RBLK 256   // reduction blocks per branch for softmax stats

typedef __bf16 v16bf __attribute__((ext_vector_type(16)));
typedef float  v8f   __attribute__((ext_vector_type(8)));

// ---------------------------------------------------------------------------
// Kernel 1: scores s[b][n] = tanh(Z_b[n,:] @ W_b^T + b_b) . q   (per branch b)
// One wave = 16 rows of Z; 4 f32 WMMA accumulators cover all 64 hidden units.
// W_b is staged into LDS as bf16 once per block.
// ---------------------------------------------------------------------------
__global__ __launch_bounds__(256) void score_kernel(
    const float* __restrict__ ZT, const float* __restrict__ ZC,
    const float* __restrict__ ZF,
    const float* __restrict__ WT, const float* __restrict__ bT,
    const float* __restrict__ WC, const float* __restrict__ bC,
    const float* __restrict__ WF, const float* __restrict__ bF,
    const float* __restrict__ q,
    float* __restrict__ s,  // [3][N]
    int N)
{
  const int br = blockIdx.y;
  const float* Z    = (br == 0) ? ZT : (br == 1) ? ZC : ZF;
  const float* W    = (br == 0) ? WT : (br == 1) ? WC : WF;
  const float* bias = (br == 0) ? bT : (br == 1) ? bC : bF;
  float* sout = s + (size_t)br * (size_t)N;

  // W (64x256 f32) -> LDS as bf16 (32 KB of the 320 KB/WGP pool)
  __shared__ __align__(32) __bf16 Wl[HH * DD];

  const int tid = threadIdx.x;
  for (int i = tid; i < HH * DD / 2; i += 256) {
    float2 f = reinterpret_cast<const float2*>(W)[i];
    Wl[2 * i]     = (__bf16)f.x;
    Wl[2 * i + 1] = (__bf16)f.y;
  }
  __syncthreads();

  const int wave   = tid >> 5;
  const int lane   = tid & 31;
  const int ntiles = (N + 15) >> 4;
  const int tile   = blockIdx.x * 8 + wave;
  if (tile >= ntiles) return;           // whole-wave uniform exit (EXEC stays all-1 for WMMA)

  const int n16 = lane & 15;            // A: row-in-tile ; B/C: column (hidden unit % 16)
  const int hh  = lane >> 4;            // lane-half selector for K sub-runs

  // A-fragment source row (clamped for a possible ragged last tile; those rows
  // are computed but never stored)
  int zr = tile * 16 + n16;
  if (zr > N - 1) zr = N - 1;
  const float* zrow = Z + (size_t)zr * DD;

  v8f acc0 = {}, acc1 = {}, acc2 = {}, acc3 = {};

#pragma unroll
  for (int ks = 0; ks < 8; ++ks) {
    // ---- A fragment (16x32 bf16): lane half 0 holds K=base+0..15,
    //      lane half 1 holds K=base+16..31 (contiguous, matching B)
    const int kbase = ks * 32 + hh * 16;
    const float4* zp  = reinterpret_cast<const float4*>(zrow + kbase);
    const float4* zp2 = reinterpret_cast<const float4*>(zrow + kbase + 8);
    float4 p0 = zp[0],  p1 = zp[1];
    float4 p2 = zp2[0], p3 = zp2[1];
    v16bf A;
    A[0]  = (__bf16)p0.x; A[1]  = (__bf16)p0.y; A[2]  = (__bf16)p0.z; A[3]  = (__bf16)p0.w;
    A[4]  = (__bf16)p1.x; A[5]  = (__bf16)p1.y; A[6]  = (__bf16)p1.z; A[7]  = (__bf16)p1.w;
    A[8]  = (__bf16)p2.x; A[9]  = (__bf16)p2.y; A[10] = (__bf16)p2.z; A[11] = (__bf16)p2.w;
    A[12] = (__bf16)p3.x; A[13] = (__bf16)p3.y; A[14] = (__bf16)p3.z; A[15] = (__bf16)p3.w;

    // ---- B fragments (32x16 bf16 = W^T tile): lane holds column n16,
    //      K = ks*32 + hh*16 .. +15, contiguous along W's row in LDS.
    const int kb = ks * 32 + hh * 16;
    const v16bf B0 = *reinterpret_cast<const v16bf*>(&Wl[(0 * 16 + n16) * DD + kb]);
    const v16bf B1 = *reinterpret_cast<const v16bf*>(&Wl[(1 * 16 + n16) * DD + kb]);
    const v16bf B2 = *reinterpret_cast<const v16bf*>(&Wl[(2 * 16 + n16) * DD + kb]);
    const v16bf B3 = *reinterpret_cast<const v16bf*>(&Wl[(3 * 16 + n16) * DD + kb]);

    acc0 = __builtin_amdgcn_wmma_f32_16x16x32_bf16(false, A, false, B0, (short)0, acc0, false, false);
    acc1 = __builtin_amdgcn_wmma_f32_16x16x32_bf16(false, A, false, B1, (short)0, acc1, false, false);
    acc2 = __builtin_amdgcn_wmma_f32_16x16x32_bf16(false, A, false, B2, (short)0, acc2, false, false);
    acc3 = __builtin_amdgcn_wmma_f32_16x16x32_bf16(false, A, false, B3, (short)0, acc3, false, false);
  }

  // ---- epilogue: t[r] = sum over 64 hidden units of tanh(acc + bias) * q
  float bb0 = bias[0 * 16 + n16], bb1 = bias[1 * 16 + n16];
  float bb2 = bias[2 * 16 + n16], bb3 = bias[3 * 16 + n16];
  float qq0 = q[0 * 16 + n16],    qq1 = q[1 * 16 + n16];
  float qq2 = q[2 * 16 + n16],    qq3 = q[3 * 16 + n16];

  float t[8];
#pragma unroll
  for (int r = 0; r < 8; ++r) {
    t[r] = tanhf(acc0[r] + bb0) * qq0 + tanhf(acc1[r] + bb1) * qq1 +
           tanhf(acc2[r] + bb2) * qq2 + tanhf(acc3[r] + bb3) * qq3;
  }
  // reduce across the 16 lanes of each half-wave (columns 0..15 of each group)
#pragma unroll
  for (int mask = 1; mask < 16; mask <<= 1) {
#pragma unroll
    for (int r = 0; r < 8; ++r) t[r] += __shfl_xor(t[r], mask, 32);
  }
  // C/D layout: VGPR r, lanes 0-15 -> row r; lanes 16-31 -> row r+8
  if (n16 == 0) {
    const int rowbase = tile * 16 + hh * 8;
#pragma unroll
    for (int r = 0; r < 8; ++r) {
      if (rowbase + r < N) sout[rowbase + r] = t[r];
    }
  }
}

// ---------------------------------------------------------------------------
// Online (max, sum-exp) combine for numerically stable global softmax.
// ---------------------------------------------------------------------------
__device__ __forceinline__ void ml_combine(float& m, float& l, float m2, float l2) {
  float M = fmaxf(m, m2);
  if (M == -INFINITY) { m = M; l = 0.f; return; }
  l = l * expf(m - M) + l2 * expf(m2 - M);
  m = M;
}

// Kernel 2: per-(branch, block) partial (max, sum-exp) over scores.
__global__ __launch_bounds__(256) void stats_partial_kernel(
    const float* __restrict__ s, float2* __restrict__ part, int N)
{
  const int br  = blockIdx.y;
  const int tid = threadIdx.x;
  const float* sb = s + (size_t)br * (size_t)N;

  float m = -INFINITY, l = 0.f;
  for (int i = blockIdx.x * 256 + tid; i < N; i += RBLK * 256) {
    float v = sb[i];
    float M = fmaxf(m, v);
    l = l * expf(m - M) + expf(v - M);
    m = M;
  }

  __shared__ float sm[256];
  __shared__ float sl[256];
  sm[tid] = m; sl[tid] = l;
  __syncthreads();
  for (int off = 128; off > 0; off >>= 1) {
    if (tid < off) {
      float mm = sm[tid], ll = sl[tid];
      ml_combine(mm, ll, sm[tid + off], sl[tid + off]);
      sm[tid] = mm; sl[tid] = ll;
    }
    __syncthreads();
  }
  if (tid == 0) part[(size_t)br * RBLK + blockIdx.x] = make_float2(sm[0], sl[0]);
}

// Kernel 3: one block per branch -> final (max, sum-exp).
__global__ __launch_bounds__(RBLK) void stats_final_kernel(
    const float2* __restrict__ part, float2* __restrict__ stats)
{
  const int br  = blockIdx.x;
  const int tid = threadIdx.x;
  float2 p = part[(size_t)br * RBLK + tid];

  __shared__ float sm[RBLK];
  __shared__ float sl[RBLK];
  sm[tid] = p.x; sl[tid] = p.y;
  __syncthreads();
  for (int off = RBLK / 2; off > 0; off >>= 1) {
    if (tid < off) {
      float mm = sm[tid], ll = sl[tid];
      ml_combine(mm, ll, sm[tid + off], sl[tid + off]);
      sm[tid] = mm; sl[tid] = ll;
    }
    __syncthreads();
  }
  if (tid == 0) stats[br] = make_float2(sm[0], sl[0]);
}

// ---------------------------------------------------------------------------
// Kernel 4: out[n,:] = a_T[n]*Z_T[n,:] + a_C[n]*Z_C[n,:] + a_F[n]*Z_F[n,:]
// 64 threads per row, float4 per thread (256 f32 per row). Memory-bound.
// ---------------------------------------------------------------------------
__global__ __launch_bounds__(256) void out_kernel(
    const float* __restrict__ ZT, const float* __restrict__ ZC,
    const float* __restrict__ ZF,
    const float* __restrict__ s, const float2* __restrict__ stats,
    float* __restrict__ out, int N)
{
  const int row = blockIdx.x * 4 + (threadIdx.x >> 6);
  const int c   = threadIdx.x & 63;
  if (row >= N) return;

  const float2 stT = stats[0], stC = stats[1], stF = stats[2];
  const float aT = expf(s[row]                       - stT.x) / stT.y;
  const float aC = expf(s[(size_t)N + row]           - stC.x) / stC.y;
  const float aF = expf(s[2 * (size_t)N + row]       - stF.x) / stF.y;

  const size_t idx = (size_t)row * (DD / 4) + c;
  const float4 vT = reinterpret_cast<const float4*>(ZT)[idx];
  const float4 vC = reinterpret_cast<const float4*>(ZC)[idx];
  const float4 vF = reinterpret_cast<const float4*>(ZF)[idx];
  float4 o;
  o.x = aT * vT.x + aC * vC.x + aF * vF.x;
  o.y = aT * vT.y + aC * vC.y + aF * vF.y;
  o.z = aT * vT.z + aC * vC.z + aF * vF.z;
  o.w = aT * vT.w + aC * vC.w + aF * vF.w;
  reinterpret_cast<float4*>(out)[idx] = o;
}

// ---------------------------------------------------------------------------
extern "C" void kernel_launch(void* const* d_in, const int* in_sizes, int n_in,
                              void* d_out, int out_size, void* d_ws, size_t ws_size,
                              hipStream_t stream) {
  const float* ZT = (const float*)d_in[0];
  const float* ZC = (const float*)d_in[1];
  const float* ZF = (const float*)d_in[2];
  const float* WT = (const float*)d_in[3];
  const float* bT = (const float*)d_in[4];
  const float* WC = (const float*)d_in[5];
  const float* bC = (const float*)d_in[6];
  const float* WF = (const float*)d_in[7];
  const float* bF = (const float*)d_in[8];
  const float* q  = (const float*)d_in[9];

  const int N = in_sizes[0] / DD;

  // Workspace layout (floats): scores [3][N] | partials [3][RBLK] float2 | stats [3] float2
  float*  s     = (float*)d_ws;
  float2* part  = (float2*)((char*)d_ws + 3 * (size_t)N * sizeof(float));
  float2* stats = part + 3 * RBLK;

  const int ntiles = (N + 15) / 16;
  dim3 g1((ntiles + 7) / 8, 3);
  score_kernel<<<g1, 256, 0, stream>>>(ZT, ZC, ZF, WT, bT, WC, bC, WF, bF, q, s, N);

  stats_partial_kernel<<<dim3(RBLK, 3), 256, 0, stream>>>(s, part, N);
  stats_final_kernel<<<3, RBLK, 0, stream>>>(part, stats);

  out_kernel<<<dim3((N + 3) / 4), 256, 0, stream>>>(ZT, ZC, ZF, s, stats,
                                                    (float*)d_out, N);
}
